// UnconsciousnessFlow_8985071583850
// MI455X (gfx1250) — matrix operand order repaired and checked by reading, not compiled, not run on hardware
//
#include <hip/hip_runtime.h>

#define D 256
#define N_NODES 50000
#define N_EDGES 262144
#define TILE_M 32   // 2 WMMA M-tiles per block

// bf16 fragment types for CDNA5 WMMA (wave32)
typedef __attribute__((ext_vector_type(16))) __bf16 v16bf;
typedef __attribute__((ext_vector_type(8)))  float  v8f;

__device__ __forceinline__ unsigned short f32_to_bf16(float f) {
  unsigned int u = __float_as_uint(f);
  u += 0x7FFFu + ((u >> 16) & 1u);   // round-to-nearest-even
  return (unsigned short)(u >> 16);
}

// A-matrix (16x32 bf16 per K-tile) lane/element mapping per CDNA5 ISA 7.12.2:
// lanes 0-15: M=lane,    elems 0-7 = K 0-7,  elems 8-15 = K 16-23
// lanes16-31: M=lane-16, elems 0-7 = K 8-15, elems 8-15 = K 24-31
// Packed LDS layout: [mt][kt][lane][elem] -> one contiguous v16bf per lane.
__device__ __forceinline__ int packedA_idx16(int r, int c) {  // r in [0,16)
  int kt = c >> 5;
  int kp = c & 31;
  int lane = r + (((kp >> 3) & 1) << 4);
  int v = (kp & 7) | (((kp >> 4) & 1) << 3);
  return kt * 512 + lane * 16 + v;
}
__device__ __forceinline__ int packedA_idx32(int r, int c) {  // r in [0,32)
  return (r >> 4) * 4096 + packedA_idx16(r & 15, c);
}

// One 32(M) x 32(N) x 256(K) GEMM slab per wave: 4 accumulators,
// 4 x v_wmma_f32_16x16x32_bf16 per K-tile (B fragments reused across 2 M-tiles).
__device__ __forceinline__ void wave_gemm2_k256(
    const unsigned short* sA,  // [2][8][512] packed A fragments (LDS)
    const unsigned short* __restrict__ wp,  // packed bf16 weight (global, L2-hot)
    int lane, int nt0, int nt1,
    v8f& a00, v8f& a01, v8f& a10, v8f& a11) {
#pragma unroll
  for (int kt = 0; kt < 8; ++kt) {
    v16bf A0 = *(const v16bf*)(sA + kt * 512 + lane * 16);
    v16bf A1 = *(const v16bf*)(sA + 4096 + kt * 512 + lane * 16);
    if (kt < 7) {
      __builtin_prefetch((const void*)(wp + ((kt + 1) * 16 + nt0) * 512 + lane * 16), 0, 3);
    }
    v16bf B0 = *(const v16bf*)(wp + (kt * 16 + nt0) * 512 + lane * 16);
    v16bf B1 = *(const v16bf*)(wp + (kt * 16 + nt1) * 512 + lane * 16);
    a00 = __builtin_amdgcn_wmma_f32_16x16x32_bf16(false, A0, false, B0, (short)0, a00, false, false);
    a01 = __builtin_amdgcn_wmma_f32_16x16x32_bf16(false, A0, false, B1, (short)0, a01, false, false);
    a10 = __builtin_amdgcn_wmma_f32_16x16x32_bf16(false, A1, false, B0, (short)0, a10, false, false);
    a11 = __builtin_amdgcn_wmma_f32_16x16x32_bf16(false, A1, false, B1, (short)0, a11, false, false);
  }
}

// Convert 4 fp32 256x256 weights -> bf16, pre-swizzled into the B-fragment
// layout: [kt][nt][lane][elem]; h=lane>>4: K=kt*32+8h+elem (elem<8), K=kt*32+16+8h+(elem-8) (elem>=8),
// N = nt*16 + (lane&15).  (one contiguous 32-byte load per lane per tile)
__global__ __launch_bounds__(256) void pack_w_kernel(
    const float* __restrict__ w0s, const float* __restrict__ w1s,
    const float* __restrict__ w2s, const float* __restrict__ w3s,
    unsigned short* __restrict__ dst) {
  int idx = blockIdx.x * 256 + threadIdx.x;       // 0 .. 4*65536
  int wsel = idx >> 16;
  int rem  = idx & 0xFFFF;
  int tile = rem >> 9;
  int within = rem & 511;
  int lane = within >> 4;
  int v = within & 15;
  int kt = tile >> 4;
  int nt = tile & 15;
  int k = kt * 32 + ((lane >= 16) ? 8 : 0) + (v & 7) + ((v >> 3) << 4);
  int n = nt * 16 + (lane & 15);
  const float* src = (wsel == 0) ? w0s : (wsel == 1) ? w1s : (wsel == 2) ? w2s : w3s;
  dst[idx] = f32_to_bf16(src[k * D + n]);
}

__global__ __launch_bounds__(256) void edge_kernel(
    const float* __restrict__ hidden, const int* __restrict__ se,
    const float* __restrict__ edges_y, const float* __restrict__ rel_table,
    const float* __restrict__ fm_ws, const float* __restrict__ fm_b,
    const float* __restrict__ fm_ow, const float* __restrict__ fm_ob,
    const unsigned short* __restrict__ pk_w1, const float* __restrict__ b1,
    const unsigned short* __restrict__ pk_w2, const float* __restrict__ b2,
    float* __restrict__ msg_sum, float* __restrict__ cnt) {
  __shared__ int   s_vi[TILE_M];
  __shared__ int   s_vj[TILE_M];
  __shared__ int   s_rel[TILE_M];
  __shared__ float s_ey[TILE_M];
  __shared__ __align__(32) float          s_m0[TILE_M * D];    // 32 KB fp32 residual
  __shared__ __align__(32) unsigned short s_AY[2 * 8 * 512];   // 16 KB: A, then y1 (reused)

  const int t = threadIdx.x;
  const int lane = t & 31;
  const int w = t >> 5;

  if (t < TILE_M) {
    size_t e = (size_t)blockIdx.x * TILE_M + t;
    s_vi[t]  = se[e * 6 + 1];
    s_vj[t]  = se[e * 6 + 2];
    s_rel[t] = se[e * 6 + 3];
    s_ey[t]  = edges_y[e];
  }
  __syncthreads();

  if (t < TILE_M) atomicAdd(cnt + s_vj[t], 1.0f);

  // _F message: elementwise fp32 gathers (L2-resident working set)
  {
    const int c = t;
    const float w0c = fm_ws[c], w1c = fm_ws[D + c], w2c = fm_ws[2 * D + c];
    const float bb = fm_b[c], ow = fm_ow[c], ob = fm_ob[c];
#pragma unroll 4
    for (int r = 0; r < TILE_M; ++r) {
      float hvi = hidden[(size_t)s_vi[r] * D + c];
      float hvj = hidden[(size_t)s_vj[r] * D + c];
      float rr  = rel_table[(size_t)s_rel[r] * D + c];
      float t01 = hvi * rr;
      float pre = hvi * w0c + t01 * w1c + (t01 * hvj) * w2c + bb;
      float m0  = fmaxf(pre, 0.f) * ow + ob;
      s_m0[r * D + c] = m0;
      s_AY[packedA_idx32(r, c)] = f32_to_bf16(m0);
    }
  }
  __syncthreads();

  const int nt0 = 2 * w, nt1 = 2 * w + 1;
  const int ncol0 = nt0 * 16 + (lane & 15);
  const int ncol1 = nt1 * 16 + (lane & 15);
  const int mbase = (lane >> 4) * 8;

  // GEMM1: y1 = relu(m0 @ W1 + b1)
  v8f p00 = {}, p01 = {}, p10 = {}, p11 = {};
  wave_gemm2_k256(s_AY, pk_w1, lane, nt0, nt1, p00, p01, p10, p11);
  __syncthreads();   // all waves finished reading A; safe to overwrite with y1
#pragma unroll
  for (int v = 0; v < 8; ++v) {
    int r = mbase + v;
    s_AY[packedA_idx16(r, ncol0)]        = f32_to_bf16(fmaxf(p00[v] + b1[ncol0], 0.f));
    s_AY[packedA_idx16(r, ncol1)]        = f32_to_bf16(fmaxf(p01[v] + b1[ncol1], 0.f));
    s_AY[4096 + packedA_idx16(r, ncol0)] = f32_to_bf16(fmaxf(p10[v] + b1[ncol0], 0.f));
    s_AY[4096 + packedA_idx16(r, ncol1)] = f32_to_bf16(fmaxf(p11[v] + b1[ncol1], 0.f));
  }
  __syncthreads();

  // GEMM2: msg = (m0 + y1 @ W2 + b2) * edges_y ; scatter-add into msg_sum[vj]
  v8f q00 = {}, q01 = {}, q10 = {}, q11 = {};
  wave_gemm2_k256(s_AY, pk_w2, lane, nt0, nt1, q00, q01, q10, q11);
#pragma unroll
  for (int v = 0; v < 8; ++v) {
    int m0r = mbase + v;
    int m1r = 16 + mbase + v;
    float ey0 = s_ey[m0r], ey1 = s_ey[m1r];
    float v00 = (s_m0[m0r * D + ncol0] + q00[v] + b2[ncol0]) * ey0;
    float v01 = (s_m0[m0r * D + ncol1] + q01[v] + b2[ncol1]) * ey0;
    float v10 = (s_m0[m1r * D + ncol0] + q10[v] + b2[ncol0]) * ey1;
    float v11 = (s_m0[m1r * D + ncol1] + q11[v] + b2[ncol1]) * ey1;
    atomicAdd(msg_sum + (size_t)s_vj[m0r] * D + ncol0, v00);
    atomicAdd(msg_sum + (size_t)s_vj[m0r] * D + ncol1, v01);
    atomicAdd(msg_sum + (size_t)s_vj[m1r] * D + ncol0, v10);
    atomicAdd(msg_sum + (size_t)s_vj[m1r] * D + ncol1, v11);
  }
}

__global__ __launch_bounds__(256) void node_kernel(
    const float* __restrict__ hidden, const float* __restrict__ ent_emb,
    const float* __restrict__ msg_sum, const float* __restrict__ cnt,
    const float* __restrict__ fh_ws, const float* __restrict__ fh_b,
    const float* __restrict__ fh_ow, const float* __restrict__ fh_ob,
    const unsigned short* __restrict__ pk_w1, const float* __restrict__ b1,
    const unsigned short* __restrict__ pk_w2, const float* __restrict__ b2,
    float* __restrict__ out) {
  __shared__ float s_cnt[TILE_M];
  __shared__ __align__(32) float          s_m0[TILE_M * D];
  __shared__ __align__(32) unsigned short s_AY[2 * 8 * 512];

  const int t = threadIdx.x;
  const int lane = t & 31;
  const int w = t >> 5;
  const int node0 = blockIdx.x * TILE_M;

  if (t < TILE_M) {
    int nd = min(node0 + t, N_NODES - 1);
    s_cnt[t] = fmaxf(cnt[nd], 1.0f);
  }
  __syncthreads();

  // _F update: terms [a, a*h, a*e, h, e, h*e]  (tail rows clamped; stores guarded)
  {
    const int c = t;
    const float w0c = fh_ws[c],         w1c = fh_ws[D + c],     w2c = fh_ws[2 * D + c];
    const float w3c = fh_ws[3 * D + c], w4c = fh_ws[4 * D + c], w5c = fh_ws[5 * D + c];
    const float bb = fh_b[c], ow = fh_ow[c], ob = fh_ob[c];
#pragma unroll 4
    for (int r = 0; r < TILE_M; ++r) {
      size_t node = (size_t)min(node0 + r, N_NODES - 1);
      float a = msg_sum[node * D + c] / s_cnt[r];
      float h = hidden[node * D + c];
      float e = ent_emb[node * D + c];
      float pre = a * w0c + (a * h) * w1c + (a * e) * w2c + h * w3c + e * w4c + (h * e) * w5c + bb;
      float u0 = fmaxf(pre, 0.f) * ow + ob;
      s_m0[r * D + c] = u0;
      s_AY[packedA_idx32(r, c)] = f32_to_bf16(u0);
    }
  }
  __syncthreads();

  const int nt0 = 2 * w, nt1 = 2 * w + 1;
  const int ncol0 = nt0 * 16 + (lane & 15);
  const int ncol1 = nt1 * 16 + (lane & 15);
  const int mbase = (lane >> 4) * 8;

  v8f p00 = {}, p01 = {}, p10 = {}, p11 = {};
  wave_gemm2_k256(s_AY, pk_w1, lane, nt0, nt1, p00, p01, p10, p11);
  __syncthreads();
#pragma unroll
  for (int v = 0; v < 8; ++v) {
    int r = mbase + v;
    s_AY[packedA_idx16(r, ncol0)]        = f32_to_bf16(fmaxf(p00[v] + b1[ncol0], 0.f));
    s_AY[packedA_idx16(r, ncol1)]        = f32_to_bf16(fmaxf(p01[v] + b1[ncol1], 0.f));
    s_AY[4096 + packedA_idx16(r, ncol0)] = f32_to_bf16(fmaxf(p10[v] + b1[ncol0], 0.f));
    s_AY[4096 + packedA_idx16(r, ncol1)] = f32_to_bf16(fmaxf(p11[v] + b1[ncol1], 0.f));
  }
  __syncthreads();

  v8f q00 = {}, q01 = {}, q10 = {}, q11 = {};
  wave_gemm2_k256(s_AY, pk_w2, lane, nt0, nt1, q00, q01, q10, q11);
#pragma unroll
  for (int v = 0; v < 8; ++v) {
    int m0r = mbase + v;
    int m1r = 16 + mbase + v;
    size_t nA = (size_t)(node0 + m0r);
    size_t nB = (size_t)(node0 + m1r);
    if ((int)nA < N_NODES) {
      out[nA * D + ncol0] = hidden[nA * D + ncol0] + s_m0[m0r * D + ncol0] + q00[v] + b2[ncol0];
      out[nA * D + ncol1] = hidden[nA * D + ncol1] + s_m0[m0r * D + ncol1] + q01[v] + b2[ncol1];
    }
    if ((int)nB < N_NODES) {
      out[nB * D + ncol0] = hidden[nB * D + ncol0] + s_m0[m1r * D + ncol0] + q10[v] + b2[ncol0];
      out[nB * D + ncol1] = hidden[nB * D + ncol1] + s_m0[m1r * D + ncol1] + q11[v] + b2[ncol1];
    }
  }
}

extern "C" void kernel_launch(void* const* d_in, const int* in_sizes, int n_in,
                              void* d_out, int out_size, void* d_ws, size_t ws_size,
                              hipStream_t stream) {
  (void)in_sizes; (void)n_in; (void)out_size; (void)ws_size;
  const float* hidden    = (const float*)d_in[0];
  const int*   se        = (const int*)  d_in[1];
  const float* edges_y   = (const float*)d_in[2];
  const float* rel_table = (const float*)d_in[3];
  const float* ent_emb   = (const float*)d_in[4];
  const float* fm_ws     = (const float*)d_in[5];
  const float* fm_b      = (const float*)d_in[6];
  const float* fm_ow     = (const float*)d_in[7];
  const float* fm_ob     = (const float*)d_in[8];
  const float* gm_w1     = (const float*)d_in[9];
  const float* gm_b1     = (const float*)d_in[10];
  const float* gm_w2     = (const float*)d_in[11];
  const float* gm_b2     = (const float*)d_in[12];
  const float* fh_ws     = (const float*)d_in[13];
  const float* fh_b      = (const float*)d_in[14];
  const float* fh_ow     = (const float*)d_in[15];
  const float* fh_ob     = (const float*)d_in[16];
  const float* gh_w1     = (const float*)d_in[17];
  const float* gh_b1     = (const float*)d_in[18];
  const float* gh_w2     = (const float*)d_in[19];
  const float* gh_b2     = (const float*)d_in[20];

  const size_t MSG_BYTES = (size_t)N_NODES * D * sizeof(float);   // 51,200,000
  const size_t CNT_BYTES = (size_t)N_NODES * sizeof(float);       //    200,000

  char* ws = (char*)d_ws;
  float* msg_sum = (float*)ws;
  float* cnt     = (float*)(ws + MSG_BYTES);
  unsigned short* wpack = (unsigned short*)(ws + MSG_BYTES + CNT_BYTES);
  unsigned short* pk_gm1 = wpack;
  unsigned short* pk_gm2 = wpack + 1 * 65536;
  unsigned short* pk_gh1 = wpack + 2 * 65536;
  unsigned short* pk_gh2 = wpack + 3 * 65536;

  hipMemsetAsync(msg_sum, 0, MSG_BYTES + CNT_BYTES, stream);

  pack_w_kernel<<<1024, 256, 0, stream>>>(gm_w1, gm_w2, gh_w1, gh_w2, wpack);

  edge_kernel<<<N_EDGES / TILE_M, 256, 0, stream>>>(
      hidden, se, edges_y, rel_table,
      fm_ws, fm_b, fm_ow, fm_ob,
      pk_gm1, gm_b1, pk_gm2, gm_b2,
      msg_sum, cnt);

  node_kernel<<<(N_NODES + TILE_M - 1) / TILE_M, 256, 0, stream>>>(
      hidden, ent_emb, msg_sum, cnt,
      fh_ws, fh_b, fh_ow, fh_ob,
      pk_gh1, gh_b1, pk_gh2, gh_b2,
      (float*)d_out);
}
